// TriangleMultiplication_9208409882903
// MI455X (gfx1250) — compile-verified, matrix operand order checked by reading, not run on hardware
//
#include <hip/hip_runtime.h>

// ---------------------------------------------------------------------------
// TriangleMultiplicationOutgoing for MI455X (gfx1250, wave32, WMMA bf16)
//   N = 512, C_z = 128, C_int = 128
// All GEMMs via v_wmma_f32_16x16x32_bf16 (fp32 accumulate).
// Triangle kernel stages operands into LDS with global_load_async_to_lds_b128
// (ASYNCcnt-tracked CDNA5 async path), double-buffered against the WMMA chain.
// ---------------------------------------------------------------------------

typedef __bf16 bf16_t;
typedef __attribute__((ext_vector_type(16))) __bf16 v16bf;
typedef __attribute__((ext_vector_type(8)))  __bf16 v8bf;
typedef __attribute__((ext_vector_type(8)))  float  v8f;

#define LN_EPS 1e-5f
#define NRES 512
#define CCH  128
#define NTOK (NRES * NRES)          // 262144 tokens
#define LDS_STRIDE 136              // 128 + 8 halves pad (272B row -> no bank conflicts)
#define WMAT_STRIDE 16384           // halves per packed 128x128 matrix (8*4*32*16)
#define TRI_PITCH 40                // LDS row pitch (halves) for staged GEMM tiles

__device__ __forceinline__ float sigmoidf(float x) {
    return 1.0f / (1.0f + __expf(-x));
}

// Load a 16x32 bf16 A/B fragment: per-lane two contiguous 8-half (16B) chunks.
__device__ __forceinline__ v16bf load_frag(const bf16_t* p0, const bf16_t* p1) {
    union { v16bf v; v8bf h[2]; } u;
    u.h[0] = *(const v8bf*)p0;
    u.h[1] = *(const v8bf*)p1;
    return u.v;
}

__device__ __forceinline__ v8f wmma_bf16(v16bf a, v16bf b, v8f c) {
    return __builtin_amdgcn_wmma_f32_16x16x32_bf16(
        /*neg_a=*/false, a, /*neg_b=*/false, b,
        /*c_mod=*/(short)0, c, /*reuse_a=*/false, /*reuse_b=*/false);
}

// Async DMA: 16 bytes per lane, global -> LDS, tracked by ASYNCcnt.
__device__ __forceinline__ void async_copy_b128(uint32_t lds_byte_off, const bf16_t* gptr) {
    uint64_t ga = (uint64_t)(uintptr_t)gptr;
    asm volatile("global_load_async_to_lds_b128 %0, %1, off"
                 :: "v"(lds_byte_off), "v"(ga) : "memory");
}

__device__ __forceinline__ void wait_asynccnt0() {
    asm volatile("s_wait_asynccnt 0x0" ::: "memory");
}

// ---------------------------------------------------------------------------
// Kernel 0: repack six fp32 128x128 weight matrices into bf16 B-fragment order.
// Packed layout (halves): [mat][nt(8)][kstep(4)][lane(32)][idx(16)]
//   n = nt*16 + (lane&15)
//   k = kstep*32 + (idx>=8 ? 16 : 0) + (lane>=16 ? 8 : 0) + (idx&7)
// matching the ISA 16-bit A/B per-lane K striping.
// ---------------------------------------------------------------------------
__global__ __launch_bounds__(256) void k_pack_weights(
    const float* __restrict__ w0, const float* __restrict__ w1,
    const float* __restrict__ w2, const float* __restrict__ w3,
    const float* __restrict__ w4, const float* __restrict__ w5,
    bf16_t* __restrict__ wpk)
{
    int gid  = blockIdx.x * 256 + threadIdx.x;   // 6*16384 = 98304 total
    int mat  = gid >> 14;
    int rem  = gid & 16383;
    int nt   = rem >> 11;
    int rem2 = rem & 2047;
    int ks   = rem2 >> 9;
    int rem3 = rem2 & 511;
    int lane = rem3 >> 4;
    int idx  = rem3 & 15;
    int n = nt * 16 + (lane & 15);
    int k = ks * 32 + ((idx & 8) ? 16 : 0) + ((lane & 16) ? 8 : 0) + (idx & 7);
    const float* W = (mat == 0) ? w0 : (mat == 1) ? w1 : (mat == 2) ? w2
                   : (mat == 3) ? w3 : (mat == 4) ? w4 : w5;
    wpk[gid] = (bf16_t)W[k * CCH + n];
}

// ---------------------------------------------------------------------------
// Kernel 1: fused input LayerNorm + 4 projections (left/right proj & gates).
// One workgroup (256 thr = 8 waves) per 16-token tile.
//   Phase A: LN in fp32, write bf16 A-tile [16][136] to LDS.
//   Phase B: wave w owns output-channel block c0=16w; 4 WMMA chains over K=128.
// Epilogue: mask*(proj+b)*sigmoid(gproj+bg), store bf16 as [c][token]
//   -> one contiguous 16B store per lane per matrix; unit-stride K for kernel 2.
// ---------------------------------------------------------------------------
__global__ __launch_bounds__(256) void k_ln_proj(
    const float* __restrict__ left_act, const float* __restrict__ left_mask,
    const float* __restrict__ ln_s,     const float* __restrict__ ln_o,
    const float* __restrict__ b_left,   const float* __restrict__ b_right,
    const float* __restrict__ b_lgate,  const float* __restrict__ b_rgate,
    const bf16_t* __restrict__ wpk,
    bf16_t* __restrict__ Lbuf, bf16_t* __restrict__ Rbuf)
{
    __shared__ bf16_t actLds[16 * LDS_STRIDE];
    __shared__ float  maskLds[16];

    const int tid  = threadIdx.x;
    const int lane = tid & 31;
    const int wv   = tid >> 5;
    const int l16  = lane & 15;
    const int half = lane >> 4;               // 0 or 1
    const int koff = half ? 8 : 0;
    const int r0   = blockIdx.x * 16;

    // ---- Phase A: LayerNorm (wave wv handles rows 2wv+half; 8 ch per lane)
    {
        int row   = 2 * wv + half;
        int cpart = l16 * 8;
        const float* src = left_act + (size_t)(r0 + row) * CCH + cpart;
        float x[8];
        float s = 0.f, s2 = 0.f;
        #pragma unroll
        for (int j = 0; j < 8; ++j) { x[j] = src[j]; s += x[j]; s2 += x[j] * x[j]; }
        #pragma unroll
        for (int m = 1; m < 16; m <<= 1) {     // reduce within each 16-lane half
            s  += __shfl_xor(s,  m, 32);
            s2 += __shfl_xor(s2, m, 32);
        }
        float mu  = s * (1.f / CCH);
        float inv = rsqrtf(s2 * (1.f / CCH) - mu * mu + LN_EPS);
        #pragma unroll
        for (int j = 0; j < 8; ++j) {
            float v = (x[j] - mu) * inv * ln_s[cpart + j] + ln_o[cpart + j];
            actLds[row * LDS_STRIDE + cpart + j] = (bf16_t)v;
        }
        if (l16 == 0) maskLds[row] = left_mask[r0 + row];
    }
    __syncthreads();

    // ---- Phase B: 4 WMMA GEMM chains, K = 128 (4 steps of 32)
    const int c0 = wv * 16;
    v8f pl = {}, pr = {}, gl = {}, gr = {};
    #pragma unroll
    for (int ks = 0; ks < 4; ++ks) {
        const bf16_t* ap = actLds + l16 * LDS_STRIDE + ks * 32 + koff;
        v16bf a = load_frag(ap, ap + 16);
        const bf16_t* wb = wpk + ((size_t)((wv * 4 + ks) * 32 + lane)) * 16;
        v16bf b0 = load_frag(wb,                   wb + 8);
        v16bf b1 = load_frag(wb + 1 * WMAT_STRIDE, wb + 1 * WMAT_STRIDE + 8);
        v16bf b2 = load_frag(wb + 2 * WMAT_STRIDE, wb + 2 * WMAT_STRIDE + 8);
        v16bf b3 = load_frag(wb + 3 * WMAT_STRIDE, wb + 3 * WMAT_STRIDE + 8);
        pl = wmma_bf16(a, b0, pl);
        pr = wmma_bf16(a, b1, pr);
        gl = wmma_bf16(a, b2, gl);
        gr = wmma_bf16(a, b3, gr);
    }

    // ---- Epilogue: gate + mask, pack 8 rows -> one 16B store per lane
    float bl  = b_left[c0 + l16];
    float br_ = b_right[c0 + l16];
    float bgl = b_lgate[c0 + l16];
    float bgr = b_rgate[c0 + l16];
    union { v8bf v; bf16_t e[8]; } lp, rp;
    #pragma unroll
    for (int vi = 0; vi < 8; ++vi) {
        int   m  = koff + vi;                 // D layout: lanes<16 -> M=0..7, else 8..15
        float mk = maskLds[m];
        lp.e[vi] = (bf16_t)(mk * (pl[vi] + bl)  * sigmoidf(gl[vi] + bgl));
        rp.e[vi] = (bf16_t)(mk * (pr[vi] + br_) * sigmoidf(gr[vi] + bgr));
    }
    size_t base = (size_t)(c0 + l16) * NTOK + r0 + koff;
    *(v8bf*)(Lbuf + base) = lp.v;
    *(v8bf*)(Rbuf + base) = rp.v;
}

// ---------------------------------------------------------------------------
// Kernel 2: triangle contraction  out[i,j,c] = sum_k L[c,i,k] * R[c,j,k]
// 128 independent 512x512x512 bf16 GEMMs (L+R = 128MB -> L2-resident).
// Workgroup: 64(i) x 256(j) per channel; 8 waves as 2x4, wave tile 32x64
//   -> 2 A-frags x 4 B-frags -> 8 WMMAs per K-step per wave.
// K-step operand blocks (A:64x32, B:256x32) staged to LDS with
// global_load_async_to_lds_b128, double-buffered (DMA of step k+1 overlaps
// the 64 WMMAs of step k). LDS rows padded to 40 halves (80B) so the 16-lane
// ds_load_b128 fragment reads are bank-conflict-free.
// ---------------------------------------------------------------------------
__global__ __launch_bounds__(256) void k_triangle(
    const bf16_t* __restrict__ Lbuf, const bf16_t* __restrict__ Rbuf,
    bf16_t* __restrict__ Tri)
{
    __shared__ bf16_t Ash[2][64 * TRI_PITCH];    // 2 x 5120 B
    __shared__ bf16_t Bsh[2][256 * TRI_PITCH];   // 2 x 20480 B

    const int tid  = threadIdx.x;
    const int lane = tid & 31;
    const int wv   = tid >> 5;
    const int l16  = lane & 15;
    const int koff = (lane & 16) ? 8 : 0;

    const int c  = blockIdx.z;
    const int i0 = blockIdx.x * 64;
    const int j0 = blockIdx.y * 256;
    const int wm = wv & 1;                       // i half within WG tile
    const int wn = wv >> 1;                      // j quarter within WG tile

    const bf16_t* Lc = Lbuf + (size_t)c * NTOK;
    const bf16_t* Rc = Rbuf + (size_t)c * NTOK;

    const uint32_t aOff[2] = { (uint32_t)(uintptr_t)&Ash[0][0],
                               (uint32_t)(uintptr_t)&Ash[1][0] };
    const uint32_t bOff[2] = { (uint32_t)(uintptr_t)&Bsh[0][0],
                               (uint32_t)(uintptr_t)&Bsh[1][0] };

    // Stage one K-step: A 64x32 (1 chunk/thread), B 256x32 (4 chunks/thread).
    auto stage = [&](int buf, int k0) {
        int ar = tid >> 2, as = tid & 3;
        async_copy_b128(aOff[buf] + (uint32_t)(ar * (TRI_PITCH * 2) + as * 16),
                        Lc + (size_t)(i0 + ar) * NRES + k0 + as * 8);
        #pragma unroll
        for (int s = 0; s < 4; ++s)
            async_copy_b128(bOff[buf] + (uint32_t)(tid * (TRI_PITCH * 2) + s * 16),
                            Rc + (size_t)(j0 + tid) * NRES + k0 + s * 8);
    };

    v8f acc[2][4] = {};
    stage(0, 0);

    for (int ks = 0; ks < NRES / 32; ++ks) {
        const int cur = ks & 1;
        wait_asynccnt0();                    // own wave's DMA for `cur` complete
        __syncthreads();                     // everyone's complete; prev reads drained
        if (ks + 1 < NRES / 32)
            stage(1 - cur, (ks + 1) * 32);   // DMA next step, overlaps WMMAs below

        v16bf af[2], bf[4];
        #pragma unroll
        for (int a = 0; a < 2; ++a) {
            const bf16_t* p = &Ash[cur][(wm * 32 + a * 16 + l16) * TRI_PITCH + koff];
            af[a] = load_frag(p, p + 16);
        }
        #pragma unroll
        for (int t = 0; t < 4; ++t) {
            const bf16_t* p = &Bsh[cur][(wn * 64 + t * 16 + l16) * TRI_PITCH + koff];
            bf[t] = load_frag(p, p + 16);
        }
        #pragma unroll
        for (int a = 0; a < 2; ++a)
            #pragma unroll
            for (int t = 0; t < 4; ++t)
                acc[a][t] = wmma_bf16(af[a], bf[t], acc[a][t]);
    }

    #pragma unroll
    for (int a = 0; a < 2; ++a)
        #pragma unroll
        for (int t = 0; t < 4; ++t)
            #pragma unroll
            for (int vi = 0; vi < 8; ++vi) {
                int i = i0 + wm * 32 + a * 16 + koff + vi;
                int j = j0 + wn * 64 + t * 16 + l16;
                Tri[((size_t)i * NRES + j) * CCH + c] = (bf16_t)acc[a][t][vi];
            }
}

// ---------------------------------------------------------------------------
// Kernel 3: LN over C_int + out-projection, plus gate = sigmoid(LN(in)@Wg + bg).
// Input LN is recomputed here (cheap) to avoid a 128MB fp32 intermediate.
// Same 16-token-tile structure as kernel 1; two WMMA chains per wave.
// ---------------------------------------------------------------------------
__global__ __launch_bounds__(256) void k_out(
    const float*  __restrict__ left_act,
    const float*  __restrict__ ln_in_s, const float* __restrict__ ln_in_o,
    const bf16_t* __restrict__ Tri,
    const float*  __restrict__ ln_c_s,  const float* __restrict__ ln_c_o,
    const float*  __restrict__ b_out,   const float* __restrict__ b_gate,
    const bf16_t* __restrict__ wpk,
    float* __restrict__ out)
{
    __shared__ bf16_t triLds[16 * LDS_STRIDE];
    __shared__ bf16_t actLds[16 * LDS_STRIDE];

    const int tid  = threadIdx.x;
    const int lane = tid & 31;
    const int wv   = tid >> 5;
    const int l16  = lane & 15;
    const int half = lane >> 4;
    const int koff = half ? 8 : 0;
    const int r0   = blockIdx.x * 16;

    // ---- Phase A: two LayerNorms (tri channels + recomputed input LN)
    {
        int row   = 2 * wv + half;
        int cpart = l16 * 8;

        // LN over C_int of the triangle output (bf16 in, fp32 stats)
        v8bf tv = *(const v8bf*)(Tri + (size_t)(r0 + row) * CCH + cpart);
        float x[8];
        float s = 0.f, s2 = 0.f;
        #pragma unroll
        for (int j = 0; j < 8; ++j) { x[j] = (float)tv[j]; s += x[j]; s2 += x[j] * x[j]; }
        #pragma unroll
        for (int m = 1; m < 16; m <<= 1) { s += __shfl_xor(s, m, 32); s2 += __shfl_xor(s2, m, 32); }
        float mu  = s * (1.f / CCH);
        float inv = rsqrtf(s2 * (1.f / CCH) - mu * mu + LN_EPS);
        #pragma unroll
        for (int j = 0; j < 8; ++j) {
            float v = (x[j] - mu) * inv * ln_c_s[cpart + j] + ln_c_o[cpart + j];
            triLds[row * LDS_STRIDE + cpart + j] = (bf16_t)v;
        }

        // Recompute input LN (for the output gate)
        const float* src = left_act + (size_t)(r0 + row) * CCH + cpart;
        float y[8];
        s = 0.f; s2 = 0.f;
        #pragma unroll
        for (int j = 0; j < 8; ++j) { y[j] = src[j]; s += y[j]; s2 += y[j] * y[j]; }
        #pragma unroll
        for (int m = 1; m < 16; m <<= 1) { s += __shfl_xor(s, m, 32); s2 += __shfl_xor(s2, m, 32); }
        mu  = s * (1.f / CCH);
        inv = rsqrtf(s2 * (1.f / CCH) - mu * mu + LN_EPS);
        #pragma unroll
        for (int j = 0; j < 8; ++j) {
            float v = (y[j] - mu) * inv * ln_in_s[cpart + j] + ln_in_o[cpart + j];
            actLds[row * LDS_STRIDE + cpart + j] = (bf16_t)v;
        }
    }
    __syncthreads();

    // ---- Phase B: out-projection & gate GEMMs (packed mats 4 and 5)
    const int c0 = wv * 16;
    v8f ao = {}, ag = {};
    #pragma unroll
    for (int ks = 0; ks < 4; ++ks) {
        const bf16_t* tp = triLds + l16 * LDS_STRIDE + ks * 32 + koff;
        const bf16_t* ip = actLds + l16 * LDS_STRIDE + ks * 32 + koff;
        v16bf at = load_frag(tp, tp + 16);
        v16bf ai = load_frag(ip, ip + 16);
        const bf16_t* wb = wpk + ((size_t)((wv * 4 + ks) * 32 + lane)) * 16;
        v16bf bo = load_frag(wb + 4 * WMAT_STRIDE, wb + 4 * WMAT_STRIDE + 8);
        v16bf bg = load_frag(wb + 5 * WMAT_STRIDE, wb + 5 * WMAT_STRIDE + 8);
        ao = wmma_bf16(at, bo, ao);
        ag = wmma_bf16(ai, bg, ag);
    }

    float bo = b_out[c0 + l16];
    float bg = b_gate[c0 + l16];
    #pragma unroll
    for (int vi = 0; vi < 8; ++vi) {
        int m = koff + vi;
        out[(size_t)(r0 + m) * CCH + c0 + l16] =
            (ao[vi] + bo) * sigmoidf(ag[vi] + bg);
    }
}

// ---------------------------------------------------------------------------
extern "C" void kernel_launch(void* const* d_in, const int* in_sizes, int n_in,
                              void* d_out, int out_size, void* d_ws, size_t ws_size,
                              hipStream_t stream)
{
    const float* left_act  = (const float*)d_in[0];
    const float* left_mask = (const float*)d_in[1];
    const float* ln_in_s   = (const float*)d_in[2];
    const float* ln_in_o   = (const float*)d_in[3];
    const float* w_left    = (const float*)d_in[4];
    const float* b_left    = (const float*)d_in[5];
    const float* w_right   = (const float*)d_in[6];
    const float* b_right   = (const float*)d_in[7];
    const float* w_lgate   = (const float*)d_in[8];
    const float* b_lgate   = (const float*)d_in[9];
    const float* w_rgate   = (const float*)d_in[10];
    const float* b_rgate   = (const float*)d_in[11];
    const float* ln_c_s    = (const float*)d_in[12];
    const float* ln_c_o    = (const float*)d_in[13];
    const float* w_out     = (const float*)d_in[14];
    const float* b_out     = (const float*)d_in[15];
    const float* w_gate    = (const float*)d_in[16];
    const float* b_gate    = (const float*)d_in[17];
    float* outp = (float*)d_out;

    char* ws = (char*)d_ws;
    const size_t MB = 1024ull * 1024ull;
    bf16_t* wpk  = (bf16_t*)ws;                        // 192 KB packed weights
    bf16_t* Lbuf = (bf16_t*)(ws + 1 * MB);             // 64 MB  [c][i*512+k]
    bf16_t* Rbuf = (bf16_t*)(ws + 65 * MB);            // 64 MB  [c][j*512+k]
    bf16_t* Tri  = (bf16_t*)(ws + 129 * MB);           // 64 MB  [i][j][c]

    k_pack_weights<<<384, 256, 0, stream>>>(w_left, w_right, w_lgate, w_rgate,
                                            w_out, w_gate, wpk);
    k_ln_proj<<<NTOK / 16, 256, 0, stream>>>(left_act, left_mask, ln_in_s, ln_in_o,
                                             b_left, b_right, b_lgate, b_rgate,
                                             wpk, Lbuf, Rbuf);
    k_triangle<<<dim3(NRES / 64, NRES / 256, CCH), 256, 0, stream>>>(Lbuf, Rbuf, Tri);
    k_out<<<NTOK / 16, 256, 0, stream>>>(left_act, ln_in_s, ln_in_o, Tri,
                                         ln_c_s, ln_c_o, b_out, b_gate, wpk, outp);
}
